// ParallelAttention_64355789964087
// MI455X (gfx1250) — matrix operand order repaired and checked by reading, not compiled, not run on hardware
//
#include <hip/hip_runtime.h>

// ---------------------------------------------------------------------------
// ParallelAttention for MI455X (gfx1250, wave32, WMMA bf16 path +
// async global->LDS staging where no conversion/transpose is needed)
// B=2, S=2048, H=4096, NH=32, HD=128
// ---------------------------------------------------------------------------

typedef __bf16 bf16;
typedef __attribute__((ext_vector_type(16))) __bf16 v16bf;
typedef __attribute__((ext_vector_type(8)))  __bf16 bf16x8;
typedef __attribute__((ext_vector_type(8)))  float  v8f;
typedef __attribute__((ext_vector_type(4)))  float  f32x4;
typedef int v4i __attribute__((vector_size(16)));   // matches async-LDS builtin

union AB { v16bf v; bf16x8 h[2]; };

#define WMMA_BF16(a, b, c) \
    __builtin_amdgcn_wmma_f32_16x16x32_bf16(false, (a), false, (b), (short)0, (c), false, false)

constexpr int Bn = 2, Sn = 2048, Hn = 4096, NHn = 32, HDn = 128;
// LDS strides (in bf16 elements): 72 (=144B) for 64-wide tiles, 136 (=272B) for
// 128-wide tiles. Both are 16B multiples (b128-aligned) and bank-conflict free
// across the 16 lanes of a half (36*M mod 64 and 68*M mod 64 distinct, M<16).
constexpr int ST64  = 72;
constexpr int ST128 = 136;

__device__ __forceinline__ bf16x8 lds8(const bf16* p) {
    return *reinterpret_cast<const bf16x8*>(p);
}

// ---- CDNA5 async global->LDS copy (ASYNCcnt-tracked), with safe fallback ----
#if defined(__AMDGCN__) && __has_builtin(__builtin_amdgcn_global_load_async_to_lds_b128)
#define HAVE_ASYNC_LDS 1
#else
#define HAVE_ASYNC_LDS 0
#endif

__device__ __forceinline__ void async_copy_b128(const bf16* g, bf16* l) {
#if HAVE_ASYNC_LDS
    __builtin_amdgcn_global_load_async_to_lds_b128(
        (__attribute__((address_space(1))) v4i*)g,
        (__attribute__((address_space(3))) v4i*)l, 0, 0);
#else
    *reinterpret_cast<bf16x8*>(l) = *reinterpret_cast<const bf16x8*>(g);
#endif
}

__device__ __forceinline__ void wait_async() {
#if defined(__AMDGCN__) && __has_builtin(__builtin_amdgcn_s_wait_asynccnt)
    __builtin_amdgcn_s_wait_asynccnt(0);
#elif HAVE_ASYNC_LDS
    asm volatile("s_wait_asynccnt 0" ::: "memory");
#endif
}

// ---------------------------------------------------------------------------
// Kernel 1: fused QKV projection.  X[4096x4096] @ {Wq|Wk|Wv} + bias.
//   which==0: Q -> ws bf16 [b,h,s,d], pre-scaled by 1/sqrt(HD)
//   which==1: K -> d_out fp32 [b,h,s,d] AND ws bf16 [b,h,s,d]
//   which==2: V -> d_out fp32 [b,h,s,d] AND ws bf16 [b,h,s,d]
// (fp32->bf16 conversion is needed during staging, so this keeps the VGPR path)
// ---------------------------------------------------------------------------
__global__ __launch_bounds__(256, 2) void qkv_wmma_kernel(
    const float* __restrict__ X,
    const float* __restrict__ Wq, const float* __restrict__ bq,
    const float* __restrict__ Wk, const float* __restrict__ bk,
    const float* __restrict__ Wv, const float* __restrict__ bv,
    bf16* __restrict__ Qbf, bf16* __restrict__ Kbf, bf16* __restrict__ Vbf,
    float* __restrict__ KtOut, float* __restrict__ VtOut)
{
    const int mt = blockIdx.x;           // 0..31  (M tiles of 128 over B*S)
    const int nt = blockIdx.y;           // 0..95  (N tiles of 128 over 3*4096)
    const int which = nt >> 5;           // 0=Q, 1=K, 2=V
    const int n0 = (nt & 31) * 128;
    const int m0 = mt * 128;
    const float* W    = (which == 0) ? Wq : (which == 1) ? Wk : Wv;
    const float* bias = (which == 0) ? bq : (which == 1) ? bk : bv;

    __shared__ __attribute__((aligned(16))) bf16 As[128 * ST64];   // X tile [m][k]
    __shared__ __attribute__((aligned(16))) bf16 Bs[128 * ST64];   // W tile [n][k]

    const int tid   = threadIdx.x;
    const int wave  = tid >> 5;
    const int lane  = tid & 31;
    const int lrow  = lane & 15;
    const int lhalf = lane >> 4;
    const int wm = wave >> 1;            // 0..3  -> 32 M rows
    const int wn = wave & 1;             // 0..1  -> 64 N cols

    v8f acc[2][4];
    #pragma unroll
    for (int i = 0; i < 2; i++)
        #pragma unroll
        for (int j = 0; j < 4; j++)
            #pragma unroll
            for (int g = 0; g < 8; g++) acc[i][j][g] = 0.f;

    for (int k0 = 0; k0 < Hn; k0 += 64) {
        // ---- stage A: 128x64 fp32 -> bf16 LDS, row-major [m][k] ----
        #pragma unroll
        for (int i = 0; i < 8; i++) {
            int idx = tid + i * 256;                 // 0..2047 float4 chunks
            int row = idx >> 4;
            int c4  = (idx & 15) * 4;
            f32x4 v = *reinterpret_cast<const f32x4*>(
                &X[(size_t)(m0 + row) * Hn + k0 + c4]);
            bf16* d = &As[row * ST64 + c4];
            d[0] = (bf16)v.x; d[1] = (bf16)v.y; d[2] = (bf16)v.z; d[3] = (bf16)v.w;
        }
        // ---- stage B: 64x128 fp32 -> bf16 LDS, transposed [n][k] ----
        #pragma unroll
        for (int i = 0; i < 8; i++) {
            int idx = tid + i * 256;
            int kr  = idx >> 5;                      // 0..63
            int c4  = (idx & 31) * 4;                // 0..124
            f32x4 v = *reinterpret_cast<const f32x4*>(
                &W[(size_t)(k0 + kr) * Hn + n0 + c4]);
            Bs[(c4 + 0) * ST64 + kr] = (bf16)v.x;
            Bs[(c4 + 1) * ST64 + kr] = (bf16)v.y;
            Bs[(c4 + 2) * ST64 + kr] = (bf16)v.z;
            Bs[(c4 + 3) * ST64 + kr] = (bf16)v.w;
        }
        __syncthreads();

        #pragma unroll
        for (int ks = 0; ks < 64; ks += 32) {
            AB a[2], bb[4];
            #pragma unroll
            for (int i = 0; i < 2; i++) {            // A: interleaved half layout
                const bf16* p = &As[(wm * 32 + i * 16 + lrow) * ST64 + ks + lhalf * 8];
                a[i].h[0] = lds8(p);
                a[i].h[1] = lds8(p + 16);
            }
            #pragma unroll
            for (int j = 0; j < 4; j++) {            // B: interleaved half layout
                const bf16* p = &Bs[(wn * 64 + j * 16 + lrow) * ST64 + ks + lhalf * 8];
                bb[j].h[0] = lds8(p);
                bb[j].h[1] = lds8(p + 16);
            }
            #pragma unroll
            for (int i = 0; i < 2; i++)
                #pragma unroll
                for (int j = 0; j < 4; j++)
                    acc[i][j] = WMMA_BF16(a[i].v, bb[j].v, acc[i][j]);
        }
        __syncthreads();
    }

    // ---- epilogue: bias, scatter to [b,h,s,d] layouts ----
    const float qscale = 0.08838834764831845f;       // 1/sqrt(128)
    #pragma unroll
    for (int i = 0; i < 2; i++) {
        #pragma unroll
        for (int j = 0; j < 4; j++) {
            #pragma unroll
            for (int g = 0; g < 8; g++) {
                int m = m0 + wm * 32 + i * 16 + g + 8 * lhalf;
                int c = n0 + wn * 64 + j * 16 + lrow;
                float val = acc[i][j][g] + bias[c];
                int bb_ = m >> 11, s = m & (Sn - 1);
                int hh  = c >> 7,  d = c & (HDn - 1);
                size_t tix = ((size_t)(bb_ * NHn + hh) * Sn + s) * HDn + d;
                if (which == 0) {
                    Qbf[tix] = (bf16)(val * qscale);
                } else if (which == 1) {
                    KtOut[tix] = val;  Kbf[tix] = (bf16)val;
                } else {
                    VtOut[tix] = val;  Vbf[tix] = (bf16)val;
                }
            }
        }
    }
}

// ---------------------------------------------------------------------------
// Kernel 2: flash attention per (b, h, 128-row q tile).  kv tiles of 64.
// Each of the 8 waves owns 16 q rows -> wave-local online softmax.
// K tile staged via CDNA5 async global->LDS; V staged transposed via VGPRs.
// Output: bf16 attn result in [b*S + s][NH*HD] layout for the O-projection.
// ---------------------------------------------------------------------------
__global__ __launch_bounds__(256, 1) void flash_attn_wmma_kernel(
    const bf16* __restrict__ Qbf, const bf16* __restrict__ Kbf,
    const bf16* __restrict__ Vbf, bf16* __restrict__ AttnBf)
{
    const int qt = blockIdx.x;                       // 0..15
    const int bh = blockIdx.y;                       // 0..63
    const int b  = bh >> 5, h = bh & 31;
    const size_t base = (size_t)(b * NHn + h) * Sn * HDn;
    const bf16* Qp = Qbf + base;
    const bf16* Kp = Kbf + base;
    const bf16* Vp = Vbf + base;

    // KP: K tile [kv=64][d=128] (stride 136) ALIASED with P tile [q=128][kv=64]
    // (stride 72).  Vts: V^T tile [d=128][kv=64] (stride 72).
    __shared__ __attribute__((aligned(16))) bf16 KP [128 * ST64];  // 9216 elems
    __shared__ __attribute__((aligned(16))) bf16 Vts[128 * ST64];  // 9216 elems

    const int tid   = threadIdx.x;
    const int wave  = tid >> 5;
    const int lane  = tid & 31;
    const int lrow  = lane & 15;
    const int lhalf = lane >> 4;
    const int q0 = qt * 128;

    // Q fragments for this wave's 16 rows, straight from global (b128 loads).
    AB qf[4];
    #pragma unroll
    for (int ks = 0; ks < 4; ks++) {
        const bf16* p = &Qp[(size_t)(q0 + wave * 16 + lrow) * HDn + ks * 32 + lhalf * 8];
        qf[ks].h[0] = *reinterpret_cast<const bf16x8*>(p);
        qf[ks].h[1] = *reinterpret_cast<const bf16x8*>(p + 16);
    }

    float rmax[8], lsum[8];
    v8f Oacc[8];
    #pragma unroll
    for (int g = 0; g < 8; g++) { rmax[g] = -3.0e38f; lsum[g] = 0.f; }
    #pragma unroll
    for (int j = 0; j < 8; j++)
        #pragma unroll
        for (int g = 0; g < 8; g++) Oacc[j][g] = 0.f;

    const int nkt = 2 * qt + 2;                      // causal: kv tiles of 64
    for (int kt = 0; kt < nkt; kt++) {
        // ---- stage K [kv][d] (async DMA to LDS) and V^T [d][kv] (VGPR) ----
        #pragma unroll
        for (int i = 0; i < 4; i++) {
            int idx = tid + i * 256;                 // 0..1023 b128 chunks
            int row = idx >> 4;                      // kv row 0..63
            int c8  = (idx & 15) * 8;                // d 0..120
            async_copy_b128(&Kp[(size_t)(kt * 64 + row) * HDn + c8],
                            &KP[row * ST128 + c8]);
            bf16x8 vv = *reinterpret_cast<const bf16x8*>(
                &Vp[(size_t)(kt * 64 + row) * HDn + c8]);
            #pragma unroll
            for (int e = 0; e < 8; e++) Vts[(c8 + e) * ST64 + row] = vv[e];
        }
        wait_async();          // ASYNCcnt==0: K tile resident in LDS
        __syncthreads();

        // ---- S = Q K^T (scale pre-folded into Q) ----
        v8f Sacc[4];
        #pragma unroll
        for (int j = 0; j < 4; j++)
            #pragma unroll
            for (int g = 0; g < 8; g++) Sacc[j][g] = 0.f;
        #pragma unroll
        for (int ks = 0; ks < 4; ks++) {
            #pragma unroll
            for (int j = 0; j < 4; j++) {
                AB bb;
                const bf16* p = &KP[(j * 16 + lrow) * ST128 + ks * 32 + lhalf * 8];
                bb.h[0] = lds8(p);
                bb.h[1] = lds8(p + 16);
                Sacc[j] = WMMA_BF16(qf[ks].v, bb.v, Sacc[j]);
            }
        }

        // ---- causal mask (only tiles that can touch the diagonal) ----
        if (kt >= 2 * qt) {
            #pragma unroll
            for (int j = 0; j < 4; j++) {
                int kg = kt * 64 + j * 16 + lrow;
                #pragma unroll
                for (int g = 0; g < 8; g++) {
                    int qg = q0 + wave * 16 + g + 8 * lhalf;
                    if (kg > qg) Sacc[j][g] = -3.0e38f;
                }
            }
        }

        // ---- wave-local online softmax (rows owned exclusively) ----
        #pragma unroll
        for (int g = 0; g < 8; g++) {
            float mx = Sacc[0][g];
            #pragma unroll
            for (int j = 1; j < 4; j++) mx = fmaxf(mx, Sacc[j][g]);
            #pragma unroll
            for (int msk = 1; msk < 16; msk <<= 1)
                mx = fmaxf(mx, __shfl_xor(mx, msk, 32));
            float mnew = fmaxf(rmax[g], mx);
            float corr = __expf(rmax[g] - mnew);
            rmax[g] = mnew;
            float sum = 0.f;
            #pragma unroll
            for (int j = 0; j < 4; j++) {
                float pp = __expf(Sacc[j][g] - mnew);
                Sacc[j][g] = pp;
                sum += pp;
            }
            #pragma unroll
            for (int msk = 1; msk < 16; msk <<= 1)
                sum += __shfl_xor(sum, msk, 32);
            lsum[g] = lsum[g] * corr + sum;
            #pragma unroll
            for (int j = 0; j < 8; j++) Oacc[j][g] *= corr;
        }

        __syncthreads();   // all waves done reading K before P overwrites it

        // ---- P (bf16) into KP buffer, [q][kv] stride 72 ----
        #pragma unroll
        for (int j = 0; j < 4; j++)
            #pragma unroll
            for (int g = 0; g < 8; g++)
                KP[(wave * 16 + g + 8 * lhalf) * ST64 + j * 16 + lrow] = (bf16)Sacc[j][g];
        asm volatile("s_wait_dscnt 0" ::: "memory");  // wave reads only its own rows

        // ---- O += P V ----
        #pragma unroll
        for (int ks = 0; ks < 2; ks++) {
            AB a;
            const bf16* pa = &KP[(wave * 16 + lrow) * ST64 + ks * 32 + lhalf * 8];
            a.h[0] = lds8(pa);
            a.h[1] = lds8(pa + 16);
            #pragma unroll
            for (int j = 0; j < 8; j++) {
                AB bb;
                const bf16* pb = &Vts[(j * 16 + lrow) * ST64 + ks * 32 + lhalf * 8];
                bb.h[0] = lds8(pb);
                bb.h[1] = lds8(pb + 16);
                Oacc[j] = WMMA_BF16(a.v, bb.v, Oacc[j]);
            }
        }
        __syncthreads();   // done with P and V^T before next tile load
    }

    // ---- finalize: O /= l, store bf16 in [b*S + s][h*128 + d] ----
    #pragma unroll
    for (int g = 0; g < 8; g++) {
        float inv = 1.0f / lsum[g];
        int qg = q0 + wave * 16 + g + 8 * lhalf;
        size_t ob = ((size_t)(b * Sn + qg)) * (NHn * HDn) + (size_t)h * HDn;
        #pragma unroll
        for (int j = 0; j < 8; j++)
            AttnBf[ob + j * 16 + lrow] = (bf16)(Oacc[j][g] * inv);
    }
}

// ---------------------------------------------------------------------------
// Kernel 3: output projection.  Attn(bf16)[4096x4096] @ Wo + bo -> fp32 out.
// A tile staged via CDNA5 async global->LDS (pure bf16 copy).
// ---------------------------------------------------------------------------
__global__ __launch_bounds__(256, 2) void oproj_wmma_kernel(
    const bf16* __restrict__ A, const float* __restrict__ Wo,
    const float* __restrict__ bo, float* __restrict__ Out)
{
    const int m0 = blockIdx.x * 128;
    const int n0 = blockIdx.y * 128;

    __shared__ __attribute__((aligned(16))) bf16 As[128 * ST64];
    __shared__ __attribute__((aligned(16))) bf16 Bs[128 * ST64];

    const int tid   = threadIdx.x;
    const int wave  = tid >> 5;
    const int lane  = tid & 31;
    const int lrow  = lane & 15;
    const int lhalf = lane >> 4;
    const int wm = wave >> 1, wn = wave & 1;

    v8f acc[2][4];
    #pragma unroll
    for (int i = 0; i < 2; i++)
        #pragma unroll
        for (int j = 0; j < 4; j++)
            #pragma unroll
            for (int g = 0; g < 8; g++) acc[i][j][g] = 0.f;

    for (int k0 = 0; k0 < Hn; k0 += 64) {
        #pragma unroll
        for (int i = 0; i < 4; i++) {                // A: bf16 128x64 async DMA
            int idx = tid + i * 256;                 // 0..1023
            int row = idx >> 3;
            int c8  = (idx & 7) * 8;
            async_copy_b128(&A[(size_t)(m0 + row) * Hn + k0 + c8],
                            &As[row * ST64 + c8]);
        }
        #pragma unroll
        for (int i = 0; i < 8; i++) {                // B: fp32 64x128 -> [n][k]
            int idx = tid + i * 256;
            int kr  = idx >> 5;
            int c4  = (idx & 31) * 4;
            f32x4 v = *reinterpret_cast<const f32x4*>(
                &Wo[(size_t)(k0 + kr) * Hn + n0 + c4]);
            Bs[(c4 + 0) * ST64 + kr] = (bf16)v.x;
            Bs[(c4 + 1) * ST64 + kr] = (bf16)v.y;
            Bs[(c4 + 2) * ST64 + kr] = (bf16)v.z;
            Bs[(c4 + 3) * ST64 + kr] = (bf16)v.w;
        }
        wait_async();
        __syncthreads();

        #pragma unroll
        for (int ks = 0; ks < 64; ks += 32) {
            AB a[2], bb[4];
            #pragma unroll
            for (int i = 0; i < 2; i++) {
                const bf16* p = &As[(wm * 32 + i * 16 + lrow) * ST64 + ks + lhalf * 8];
                a[i].h[0] = lds8(p);
                a[i].h[1] = lds8(p + 16);
            }
            #pragma unroll
            for (int j = 0; j < 4; j++) {
                const bf16* p = &Bs[(wn * 64 + j * 16 + lrow) * ST64 + ks + lhalf * 8];
                bb[j].h[0] = lds8(p);
                bb[j].h[1] = lds8(p + 16);
            }
            #pragma unroll
            for (int i = 0; i < 2; i++)
                #pragma unroll
                for (int j = 0; j < 4; j++)
                    acc[i][j] = WMMA_BF16(a[i].v, bb[j].v, acc[i][j]);
        }
        __syncthreads();
    }

    #pragma unroll
    for (int i = 0; i < 2; i++)
        #pragma unroll
        for (int j = 0; j < 4; j++)
            #pragma unroll
            for (int g = 0; g < 8; g++) {
                int m = m0 + wm * 32 + i * 16 + g + 8 * lhalf;
                int c = n0 + wn * 64 + j * 16 + lrow;
                Out[(size_t)m * Hn + c] = acc[i][j][g] + bo[c];
            }
}

// ---------------------------------------------------------------------------
extern "C" void kernel_launch(void* const* d_in, const int* in_sizes, int n_in,
                              void* d_out, int out_size, void* d_ws, size_t ws_size,
                              hipStream_t stream)
{
    (void)in_sizes; (void)n_in; (void)out_size; (void)ws_size;

    const float* X  = (const float*)d_in[0];
    const float* Wq = (const float*)d_in[1];
    const float* bq = (const float*)d_in[2];
    const float* Wk = (const float*)d_in[3];
    const float* bk = (const float*)d_in[4];
    const float* Wv = (const float*)d_in[5];
    const float* bv = (const float*)d_in[6];
    const float* Wo = (const float*)d_in[7];
    const float* bo = (const float*)d_in[8];

    float* out = (float*)d_out;
    const size_t T = (size_t)Bn * NHn * Sn * HDn;    // 16,777,216 elements
    float* KtOut = out + T;                          // k transposed [b,h,s,d]
    float* VtOut = out + 2 * T;                      // v transposed [b,h,s,d]

    // Workspace (bf16): Q | K | V | Attn  -> 8*T bytes = 128 MiB total
    char* ws = (char*)d_ws;
    bf16* Qbf  = (bf16*)(ws);
    bf16* Kbf  = (bf16*)(ws + 2 * T);
    bf16* Vbf  = (bf16*)(ws + 4 * T);
    bf16* Attn = (bf16*)(ws + 6 * T);

    dim3 g1(32, 96);
    qkv_wmma_kernel<<<g1, 256, 0, stream>>>(X, Wq, bq, Wk, bk, Wv, bv,
                                            Qbf, Kbf, Vbf, KtOut, VtOut);
    dim3 g2(16, 64);
    flash_attn_wmma_kernel<<<g2, 256, 0, stream>>>(Qbf, Kbf, Vbf, Attn);

    dim3 g3(32, 32);
    oproj_wmma_kernel<<<g3, 256, 0, stream>>>(Attn, Wo, bo, out);
}
